// CapsuleLayer_31413390803000
// MI455X (gfx1250) — compile-verified, matrix operand order checked by reading, not run on hardware
//
#include <hip/hip_runtime.h>
#include <hip/hip_bf16.h>
#include <math.h>

// ---------------- problem constants (from reference) ----------------
#define B_    16
#define N_    1024
#define DIN   512
#define KD    256      // NUM_CAPS * DIM_CAPS
#define NC    16       // NUM_CAPS
#define ROUT  3
#define EPSV  1e-7f

typedef __attribute__((ext_vector_type(16))) __bf16 v16bf;
typedef __attribute__((ext_vector_type(8)))  float  v8f;
typedef __attribute__((ext_vector_type(4)))  unsigned int v4ui;
typedef __attribute__((ext_vector_type(8)))  int    v8i_t;
typedef __attribute__((ext_vector_type(4)))  int    v4i_t;

// ---- Tensor Data Mover availability (device pass only) ----
#if defined(__HIP_DEVICE_COMPILE__) && __has_builtin(__builtin_amdgcn_tensor_load_to_lds)
#define HAVE_TDM 1
#else
#define HAVE_TDM 0
#endif

#if defined(__HIP_DEVICE_COMPILE__) && __has_builtin(__builtin_amdgcn_s_wait_tensorcnt)
#define TENSOR_WAIT(n) __builtin_amdgcn_s_wait_tensorcnt((short)(n))
#elif defined(__HIP_DEVICE_COMPILE__)
#define TENSOR_WAIT(n) asm volatile("s_wait_tensorcnt %0" ::"i"(n) : "memory")
#else
#define TENSOR_WAIT(n) ((void)0)
#endif

union Frag {
  v16bf  v;
  __bf16 e[16];
  uint4  q[2];
};

// CDNA5 16-bit A/B fragment lane layout (CDNA5 ISA):
// half = lane/16; element e holds K = e + 8*half + (e>=8 ? 8 : 0)
// -> two contiguous 8-element chunks at K offsets 8*half and 16+8*half.
__device__ __forceinline__ int kmap(int e, int half) { return e + 8 * half + (e & 8); }

__device__ __forceinline__ v16bf ld_frag_2chunks(const __bf16* lo, const __bf16* hi) {
  Frag f;
  f.q[0] = *reinterpret_cast<const uint4*>(lo);
  f.q[1] = *reinterpret_cast<const uint4*>(hi);
  return f.v;
}

#if HAVE_TDM
// Issue one TDM 2-D tile load (bf16 elements) per D# spec, CDNA5 ISA.
// tile rows land contiguously in LDS: [tile_d1][tile_d0] row-major.
__device__ __forceinline__ void tdm_load_2d(unsigned lds_off, const void* gptr,
                                            unsigned tensor_d0, unsigned tensor_d1,
                                            unsigned long long stride0_elems,
                                            unsigned tile_d0, unsigned tile_d1) {
  const unsigned long long ga = (unsigned long long)(size_t)gptr;
  v4ui g0;
  g0[0] = 1u;                                        // count=1 (valid), user mode
  g0[1] = lds_off;                                   // LDS byte address
  g0[2] = (unsigned)ga;                              // global_addr[31:0]
  g0[3] = (unsigned)((ga >> 32) & 0x1FFFFFFu) | (2u << 30);  // addr[56:32] | type=2
  v8i_t g1;
  g1[0] = (int)(1u << 16);                           // workgroup_mask=0, data_size=1 (2B)
  g1[1] = (int)((tensor_d0 & 0xFFFFu) << 16);        // tensor_dim0[15:0] @ bits 63:48
  g1[2] = (int)(((tensor_d0 >> 16) & 0xFFFFu) | ((tensor_d1 & 0xFFFFu) << 16));
  g1[3] = (int)(((tensor_d1 >> 16) & 0xFFFFu) | ((tile_d0 & 0xFFFFu) << 16));
  g1[4] = (int)(tile_d1 & 0xFFFFu);                  // tile_dim1 | tile_dim2=0
  g1[5] = (int)(unsigned)(stride0_elems & 0xFFFFFFFFull);
  g1[6] = (int)(unsigned)((stride0_elems >> 32) & 0xFFFFull);  // dim1_stride=0 (2-D)
  g1[7] = 0;
  v4i_t g2 = {0, 0, 0, 0}, g3 = {0, 0, 0, 0};        // <=2-D tensor: groups 2/3 unused
#if defined(__clang_major__) && (__clang_major__ >= 23)
  v8i_t g4 = {0, 0, 0, 0, 0, 0, 0, 0};
  __builtin_amdgcn_tensor_load_to_lds(g0, g1, g2, g3, g4, 0);
#else
  __builtin_amdgcn_tensor_load_to_lds(g0, g1, g2, g3, 0);
#endif
}
#endif  // HAVE_TDM

// ---------------- kernel 1: fp32 -> bf16 convert ----------------
__global__ void cvt_bf16(const float* __restrict__ in, __bf16* __restrict__ out, int n) {
  int i = blockIdx.x * blockDim.x + threadIdx.x;
  int stride = gridDim.x * blockDim.x;
  for (; i < n; i += stride) out[i] = (__bf16)in[i];
}

// ---------------- kernel 2: H = X @ W  (bf16 in, fp32 accum, bf16 out) ----
// M = 16384, K = 512, N = 256. Tile 128x64 per WG, 8 waves in 4x2, each 32x32.
// A tile staged by TDM (double buffered, overlapped via s_wait_tensorcnt 1).
__global__ void __launch_bounds__(256)
gemm_h_bf16(const __bf16* __restrict__ X, const __bf16* __restrict__ Wm,
            __bf16* __restrict__ Hb) {
#if HAVE_TDM
  __shared__ alignas(16) __bf16 As[2][128][32];   // 16 KB, double buffered
#else
  __shared__ alignas(16) __bf16 As[1][128][32];
#endif
  __shared__ alignas(16) __bf16 Bs[64][32];       // B tile transposed [n][k]

  const int tid   = threadIdx.x;
  const int lane  = tid & 31;
  const int wave  = tid >> 5;
  const int half  = lane >> 4;
  const int lan16 = lane & 15;
  const int wm    = wave & 3;
  const int wn    = wave >> 2;
  const int mbase = blockIdx.y * 128;
  const int nbase = blockIdx.x * 64;

#if HAVE_TDM
  if (wave == 0)
    tdm_load_2d((unsigned)(size_t)&As[0][0][0], X + (size_t)mbase * DIN,
                DIN, B_ * N_, (unsigned long long)DIN, 32u, 128u);
#endif

  v8f acc[2][2] = {};

  for (int k0 = 0; k0 < DIN; k0 += 32) {
#if HAVE_TDM
    const int cur = (k0 >> 5) & 1;
#else
    const int cur = 0;
#endif
    // B tile -> regs (before barrier: global read only)
    const int kk = tid & 31, n0 = (tid >> 5) * 8;
    Frag tb;
    tb.q[0] = *reinterpret_cast<const uint4*>(Wm + (size_t)(k0 + kk) * KD + nbase + n0);
#if !HAVE_TDM
    const int r = tid >> 1, c = (tid & 1) * 16;
    const uint4* asrc =
        reinterpret_cast<const uint4*>(X + (size_t)(mbase + r) * DIN + k0 + c);
    uint4 a0 = asrc[0], a1 = asrc[1];
#endif
    __syncthreads();   // all reads of As/Bs from previous iteration complete
#pragma unroll
    for (int j = 0; j < 8; ++j) Bs[n0 + j][kk] = tb.e[j];
#if HAVE_TDM
    if (wave == 0) {
      if (k0 + 32 < DIN) {
        // prefetch next A slab into the other buffer, then wait for current
        tdm_load_2d((unsigned)(size_t)&As[cur ^ 1][0][0],
                    X + (size_t)mbase * DIN + (k0 + 32),
                    DIN, B_ * N_, (unsigned long long)DIN, 32u, 128u);
        TENSOR_WAIT(1);      // TDM retires in order: waits for the older load
      } else {
        TENSOR_WAIT(0);
      }
    }
#else
    {
      uint4* dst = reinterpret_cast<uint4*>(&As[0][(tid >> 1)][(tid & 1) * 16]);
      dst[0] = a0;
      dst[1] = a1;
    }
#endif
    if (k0 + 32 < DIN)
      __builtin_prefetch(Wm + (size_t)(k0 + 32 + kk) * KD + nbase + n0, 0, 0);
    __syncthreads();   // A(cur) + B staged

    v16bf fa[2], fb[2];
#pragma unroll
    for (int i = 0; i < 2; ++i) {
      const int row = wm * 32 + i * 16 + lan16;
      fa[i] = ld_frag_2chunks(&As[cur][row][8 * half], &As[cur][row][16 + 8 * half]);
      const int col = wn * 32 + i * 16 + lan16;
      fb[i] = ld_frag_2chunks(&Bs[col][8 * half], &Bs[col][16 + 8 * half]);
    }
#pragma unroll
    for (int i = 0; i < 2; ++i)
#pragma unroll
      for (int j = 0; j < 2; ++j)
        acc[i][j] = __builtin_amdgcn_wmma_f32_16x16x32_bf16(
            false, fa[i], false, fb[j], (short)0, acc[i][j], false, false);
  }

  // C/D layout: VGPR v, lanes 0-15 -> M=v, lanes 16-31 -> M=v+8; N = lane%16
#pragma unroll
  for (int i = 0; i < 2; ++i)
#pragma unroll
    for (int j = 0; j < 2; ++j)
#pragma unroll
      for (int v = 0; v < 8; ++v) {
        const int m = mbase + wm * 32 + i * 16 + v + 8 * half;
        const int n = nbase + wn * 32 + j * 16 + lan16;
        Hb[(size_t)m * KD + n] = (__bf16)acc[i][j][v];
      }
}

// ---------------- kernel 3: dynamic routing (one workgroup per batch) ------
__global__ void __launch_bounds__(512)
capsule_routing(const __bf16* __restrict__ Hb, float* __restrict__ blog,
                float* __restrict__ out) {
  __shared__ alignas(16) __bf16 cT[NC][N_];     // 32 KB : c transposed [k][n]
  __shared__ alignas(16) __bf16 outsm[NC][KD];  //  8 KB : out [k][d]
#if HAVE_TDM
  __shared__ alignas(16) __bf16 Hs[32][KD];     // 16 KB : H slab [n][d] via TDM
#endif
  __shared__ float ssum[NC];
  __shared__ float scal[NC];

  const int batch = blockIdx.x;
  const int tid   = threadIdx.x;
  const int lane  = tid & 31;
  const int wave  = tid >> 5;   // 0..15
  const int half  = lane >> 4;
  const int lan16 = lane & 15;

  const __bf16* H  = Hb + (size_t)batch * N_ * KD;
  float*        bl = blog + (size_t)batch * N_ * NC;

  for (int i = tid; i < N_ * NC; i += blockDim.x) bl[i] = 0.0f;  // b = 0
  __threadfence_block();
  __syncthreads();

  for (int it = 0; it < ROUT; ++it) {
    // ---- softmax over k; zero squash accumulators ----
    if (tid < NC) ssum[tid] = 0.0f;
#pragma unroll
    for (int rr = 0; rr < 2; ++rr) {
      const int n = tid * 2 + rr;
      float bv[NC];
      float mx = -1e30f;
#pragma unroll
      for (int k = 0; k < NC; ++k) { bv[k] = bl[n * NC + k]; mx = fmaxf(mx, bv[k]); }
      float sum = 0.0f;
#pragma unroll
      for (int k = 0; k < NC; ++k) { bv[k] = __expf(bv[k] - mx); sum += bv[k]; }
      const float inv = 1.0f / sum;
#pragma unroll
      for (int k = 0; k < NC; ++k) cT[k][n] = (__bf16)(bv[k] * inv);
    }
    __syncthreads();

    // ---- s = cT(16 x N_) @ H(N_ x KD); wave handles 16-wide d tile ----
    const int d0 = wave * 16;
    v8f acc = {};
    for (int kb = 0; kb < N_; kb += 32) {
#if HAVE_TDM
      __syncthreads();                    // previous Hs reads complete
      if (wave == 0) {
        tdm_load_2d((unsigned)(size_t)&Hs[0][0], H + (size_t)kb * KD,
                    KD, N_, (unsigned long long)KD, (unsigned)KD, 32u);
        TENSOR_WAIT(0);
      }
      __syncthreads();                    // Hs slab ready
#endif
      v16bf fa = ld_frag_2chunks(&cT[lan16][kb + 8 * half],
                                 &cT[lan16][kb + 16 + 8 * half]);
      Frag fb;
#if HAVE_TDM
#pragma unroll
      for (int e2 = 0; e2 < 16; ++e2)
        fb.e[e2] = Hs[kmap(e2, half)][d0 + lan16];
#else
      const __bf16* hcol = H + d0 + lan16;
#pragma unroll
      for (int e2 = 0; e2 < 16; ++e2)
        fb.e[e2] = hcol[(size_t)(kb + kmap(e2, half)) * KD];
#endif
      acc = __builtin_amdgcn_wmma_f32_16x16x32_bf16(
          false, fa, false, fb.v, (short)0, acc, false, false);
    }

    // ---- squash: ssum[k] = sum_d s[k,d]^2 via LDS float atomics ----
#pragma unroll
    for (int v = 0; v < 8; ++v)
      atomicAdd(&ssum[v + 8 * half], acc[v] * acc[v]);
    __syncthreads();
    if (tid < NC) {
      const float sq = ssum[tid];
      scal[tid] = (sq / (1.0f + sq)) * rsqrtf(sq + EPSV);
    }
    __syncthreads();
#pragma unroll
    for (int v = 0; v < 8; ++v) {
      const int k = v + 8 * half;
      const float o = scal[k] * acc[v];
      outsm[k][d0 + lan16] = (__bf16)o;
      if (it == ROUT - 1)
        out[((size_t)batch * NC + k) * KD + d0 + lan16] = o;
    }
    __syncthreads();

    // ---- agreement: b[n,k] += sum_d H[n,d] * out[k,d] ----
    if (it < ROUT - 1) {
#pragma unroll
      for (int tt = 0; tt < 4; ++tt) {
        const int mb = (tt * 16 + wave) * 16;
        v8f a2 = {};
        for (int kb = 0; kb < KD; kb += 32) {
          const __bf16* hrow = H + (size_t)(mb + lan16) * KD;
          v16bf fa  = ld_frag_2chunks(hrow + kb + 8 * half,
                                      hrow + kb + 16 + 8 * half);
          v16bf fbv = ld_frag_2chunks(&outsm[lan16][kb + 8 * half],
                                      &outsm[lan16][kb + 16 + 8 * half]);
          a2 = __builtin_amdgcn_wmma_f32_16x16x32_bf16(
              false, fa, false, fbv, (short)0, a2, false, false);
        }
#pragma unroll
        for (int v = 0; v < 8; ++v) {
          const int n = mb + v + 8 * half;
          bl[n * NC + lan16] += a2[v];
        }
      }
      __threadfence_block();
      __syncthreads();
    }
  }
}

// ---------------- launcher ----------------
extern "C" void kernel_launch(void* const* d_in, const int* in_sizes, int n_in,
                              void* d_out, int out_size, void* d_ws, size_t ws_size,
                              hipStream_t stream) {
  (void)in_sizes; (void)n_in; (void)out_size; (void)ws_size;
  const float* X = (const float*)d_in[0];   // [16,1024,512]
  const float* W = (const float*)d_in[1];   // [512,256]
  float* out = (float*)d_out;               // [16,16,256]

  // workspace layout (~26.5 MB)
  char* ws = (char*)d_ws;
  size_t off = 0;
  __bf16* Xb = (__bf16*)(ws + off); off += (size_t)B_ * N_ * DIN * sizeof(__bf16);
  __bf16* Wb = (__bf16*)(ws + off); off += (size_t)DIN * KD * sizeof(__bf16);
  __bf16* Hb = (__bf16*)(ws + off); off += (size_t)B_ * N_ * KD * sizeof(__bf16);
  float*  bl = (float*)(ws + off);  off += (size_t)B_ * N_ * NC * sizeof(float);

  cvt_bf16<<<4096, 256, 0, stream>>>(X, Xb, B_ * N_ * DIN);
  cvt_bf16<<<512, 256, 0, stream>>>(W, Wb, DIN * KD);

  dim3 ggrid(KD / 64, (B_ * N_) / 128);     // (4, 128)
  gemm_h_bf16<<<ggrid, 256, 0, stream>>>(Xb, Wb, Hb);

  capsule_routing<<<B_, 512, 0, stream>>>(Hb, bl, out);
}
